// SwinTransformerBlock_28956669510295
// MI455X (gfx1250) — matrix operand order repaired and checked by reading, not compiled, not run on hardware
//
#include <hip/hip_runtime.h>
#include <math.h>

typedef __attribute__((ext_vector_type(16))) _Float16 v16h;
typedef __attribute__((ext_vector_type(8)))  float    v8f;

#define DIM     192
#define HEADS   6
#define NTOK    343
#define NWIN    128
#define NPAD    384
#define SITES   43904      // 14*56*56 == NWIN*NTOK
#define MLPH    768

// ---------- WMMA fragment loaders (ISA 7.12.2 layouts, wave32) ----------
// A (16x32 f16): lane&15 = M row; halves 0..7 at K=8*(lane>>4)+0..7,
// halves 8..15 at K=16+8*(lane>>4)+0..7  -> two contiguous 16B runs.
__device__ __forceinline__ v16h frag_a(const _Float16* p0, int stride, int lane) {
  union { uint4 q[2]; v16h v; } f;
  const _Float16* p = p0 + (size_t)(lane & 15) * stride + ((lane >> 4) << 3);
  f.q[0] = *reinterpret_cast<const uint4*>(p);
  f.q[1] = *reinterpret_cast<const uint4*>(p + 16);
  return f.v;
}
// B (32x16 f16) staged N-major: lane&15 = N col; halves 0..7 at K=8*(lane>>4)+0..7,
// halves 8..15 at K=16+8*(lane>>4)+0..7 -> two contiguous 16B runs.
__device__ __forceinline__ v16h frag_b(const _Float16* p0, int stride, int lane) {
  union { uint4 q[2]; v16h v; } f;
  const _Float16* p = p0 + (size_t)(lane & 15) * stride + ((lane >> 4) << 3);
  f.q[0] = *reinterpret_cast<const uint4*>(p);
  f.q[1] = *reinterpret_cast<const uint4*>(p + 16);
  return f.v;
}

// ---------- generic GEMM: C[M,N] = A[M,K](f16) * W[N,K]^T(f16) + bias ----------
// MODE 0: qkv epilogue (scale q, scatter to q/k/v head-major f16 buffers)
// MODE 1/3: f32 out + bias.  MODE 2: exact GELU -> f16 out.
template <int MODE>
__global__ void __launch_bounds__(256)
gemm_kernel(const _Float16* __restrict__ A, const _Float16* __restrict__ W,
            const float* __restrict__ bias, float* __restrict__ Cf,
            _Float16* __restrict__ Ch, _Float16* __restrict__ Qo,
            _Float16* __restrict__ Ko, _Float16* __restrict__ Vo,
            int M, int N, int K) {
  __shared__ __attribute__((aligned(16))) _Float16 Al[128][32];
  __shared__ __attribute__((aligned(16))) _Float16 Bl[64][32];
  const int t = threadIdx.x, lane = t & 31, wv = t >> 5;
  const int wm = wv & 3, wn = wv >> 2;
  const int rowBase = blockIdx.y * 128, colBase = blockIdx.x * 64;
  v8f acc00 = {}, acc01 = {}, acc10 = {}, acc11 = {};
  for (int kb = 0; kb < K; kb += 32) {
    {
      const uint4* src = reinterpret_cast<const uint4*>(
          A + (size_t)(rowBase + (t >> 1)) * K + kb + ((t & 1) << 4));
      uint4* dst = reinterpret_cast<uint4*>(&Al[t >> 1][(t & 1) << 4]);
      dst[0] = src[0]; dst[1] = src[1];
    }
    if (t < 128) {
      const uint4* src = reinterpret_cast<const uint4*>(
          W + (size_t)(colBase + (t >> 1)) * K + kb + ((t & 1) << 4));
      uint4* dst = reinterpret_cast<uint4*>(&Bl[t >> 1][(t & 1) << 4]);
      dst[0] = src[0]; dst[1] = src[1];
    }
    if (kb + 32 < K)  // global_prefetch_b8 of next K tile
      __builtin_prefetch(A + (size_t)(rowBase + (t >> 1)) * K + kb + 32, 0, 0);
    __syncthreads();
    v16h a0 = frag_a(&Al[wm * 32][0],      32, lane);
    v16h a1 = frag_a(&Al[wm * 32 + 16][0], 32, lane);
    v16h b0 = frag_b(&Bl[wn * 32][0],      32, lane);
    v16h b1 = frag_b(&Bl[wn * 32 + 16][0], 32, lane);
    acc00 = __builtin_amdgcn_wmma_f32_16x16x32_f16(false, a0, false, b0, (short)0, acc00, false, false);
    acc01 = __builtin_amdgcn_wmma_f32_16x16x32_f16(false, a0, false, b1, (short)0, acc01, false, false);
    acc10 = __builtin_amdgcn_wmma_f32_16x16x32_f16(false, a1, false, b0, (short)0, acc10, false, false);
    acc11 = __builtin_amdgcn_wmma_f32_16x16x32_f16(false, a1, false, b1, (short)0, acc11, false, false);
    __syncthreads();
  }
  const int hs8 = (lane >> 4) << 3, cl = lane & 15;
#pragma unroll
  for (int ti = 0; ti < 2; ++ti) {
#pragma unroll
    for (int tj = 0; tj < 2; ++tj) {
      v8f c = (ti == 0) ? ((tj == 0) ? acc00 : acc01)
                        : ((tj == 0) ? acc10 : acc11);
      int r0  = rowBase + wm * 32 + ti * 16 + hs8;
      int col = colBase + wn * 32 + tj * 16 + cl;
#pragma unroll
      for (int e = 0; e < 8; ++e) {
        int gr = r0 + e;
        float val = c[e] + bias[col];
        if (MODE == 0) {
          int sec = col / DIM;
          int ch = col - sec * DIM;
          int head = ch >> 5, hd = ch & 31;
          int win = gr / NTOK, token = gr - win * NTOK;
          if (sec == 0) val *= 0.17677669529663688f;  // 1/sqrt(32)
          size_t idx = ((size_t)(win * HEADS + head) * NPAD + token) * 32 + hd;
          _Float16 hv = (_Float16)val;
          if (sec == 0) Qo[idx] = hv;
          else if (sec == 1) Ko[idx] = hv;
          else Vo[idx] = hv;
        } else if (MODE == 2) {
          float g = 0.5f * val * (1.0f + erff(val * 0.7071067811865475f));
          Ch[(size_t)gr * N + col] = (_Float16)g;
        } else {
          Cf[(size_t)gr * N + col] = val;
        }
      }
    }
  }
}

// ---------- flash attention: one block per (window*head, 128-row chunk) ----------
__global__ void __launch_bounds__(256)
attn_kernel(const _Float16* __restrict__ Qb, const _Float16* __restrict__ Kbf,
            const _Float16* __restrict__ Vb, const float* __restrict__ biasH,
            const float* __restrict__ mask, _Float16* __restrict__ Ob) {
  __shared__ __attribute__((aligned(16))) _Float16 Kl[352][32];   // [token][hd]
  __shared__ __attribute__((aligned(16))) _Float16 Vl[32][352];   // [hd][token]
  __shared__ __attribute__((aligned(16))) _Float16 Pl[8][16][32]; // per-wave P tile
  const int wh = blockIdx.x;
  const int win = wh / HEADS, head = wh - win * HEADS;
  const int t = threadIdx.x, lane = t & 31, wv = t >> 5;
  const _Float16* kg = Kbf + (size_t)wh * NPAD * 32;
  const _Float16* vg = Vb  + (size_t)wh * NPAD * 32;
  for (int r = t; r < 352; r += 256) {
    if (r < NTOK) {
      const uint4* src = reinterpret_cast<const uint4*>(kg + (size_t)r * 32);
      uint4* dst = reinterpret_cast<uint4*>(&Kl[r][0]);
      dst[0] = src[0]; dst[1] = src[1];
      const _Float16* vr = vg + (size_t)r * 32;
#pragma unroll
      for (int c = 0; c < 32; ++c) Vl[c][r] = vr[c];
    } else {
      uint4 z = make_uint4(0u, 0u, 0u, 0u);
      uint4* dst = reinterpret_cast<uint4*>(&Kl[r][0]);
      dst[0] = z; dst[1] = z;
#pragma unroll
      for (int c = 0; c < 32; ++c) Vl[c][r] = (_Float16)0.0f;
    }
  }
  __syncthreads();

  const int qrow0 = blockIdx.y * 128 + wv * 16;
  v16h qf;
  {
    union { uint4 q[2]; v16h v; } f;
    const _Float16* qp = Qb + (size_t)wh * NPAD * 32 +
                         (size_t)(qrow0 + (lane & 15)) * 32 + ((lane >> 4) << 3);
    f.q[0] = *reinterpret_cast<const uint4*>(qp);
    f.q[1] = *reinterpret_cast<const uint4*>(qp + 16);
    qf = f.v;
  }

  v8f O0 = {}, O1 = {};
  float mrow[8], lrow[8];
#pragma unroll
  for (int e = 0; e < 8; ++e) { mrow[e] = -2e30f; lrow[e] = 0.0f; }
  const int hs8 = (lane >> 4) << 3, cl = lane & 15;
  const float* bmBase = biasH + (size_t)head * NTOK * NTOK;
  const float* mkBase = mask + (size_t)win * NTOK * NTOK;

  for (int jj = 0; jj < 11; ++jj) {   // 11 * 32 = 352 KV columns
    const int col0 = jj * 32;
    v16h kb0 = frag_b(&Kl[col0][0],      32, lane);
    v16h kb1 = frag_b(&Kl[col0 + 16][0], 32, lane);
    v8f zc = {};
    v8f s0 = __builtin_amdgcn_wmma_f32_16x16x32_f16(false, qf, false, kb0, (short)0, zc, false, false);
    v8f s1 = __builtin_amdgcn_wmma_f32_16x16x32_f16(false, qf, false, kb1, (short)0, zc, false, false);
    float x0[8], x1[8];
    const int c0 = col0 + cl;               // always < NTOK (max 335)
    const int c1 = c0 + 16;                 // can exceed only when jj == 10
    const int c1c = (c1 < NTOK) ? c1 : (NTOK - 1);
    const float pen1 = (c1 < NTOK) ? 0.0f : -2e30f;
#pragma unroll
    for (int e = 0; e < 8; ++e) {
      int r = qrow0 + hs8 + e;
      const bool okr = (r < NTOK);
      int rc = okr ? r : (NTOK - 1);        // clamped: loads always in-bounds
      const float penr = okr ? 0.0f : -2e30f;
      const float* bm = bmBase + (size_t)rc * NTOK;
      const float* mk = mkBase + (size_t)rc * NTOK;
      // loads stay live on every path -> straight-line code + v_cndmask on pen
      x0[e] = s0[e] + (bm[c0] + mk[c0]) + penr;
      x1[e] = s1[e] + (bm[c1c] + mk[c1c]) + (penr + pen1);
    }
#pragma unroll
    for (int e = 0; e < 8; ++e) {
      float tm = fmaxf(x0[e], x1[e]);
      tm = fmaxf(tm, __shfl_xor(tm, 1, 32));
      tm = fmaxf(tm, __shfl_xor(tm, 2, 32));
      tm = fmaxf(tm, __shfl_xor(tm, 4, 32));
      tm = fmaxf(tm, __shfl_xor(tm, 8, 32));
      float mNew = fmaxf(mrow[e], tm);
      float sc = __expf(mrow[e] - mNew);
      mrow[e] = mNew;
      float p0 = __expf(x0[e] - mNew);
      float p1 = __expf(x1[e] - mNew);
      float rs = p0 + p1;
      rs += __shfl_xor(rs, 1, 32);
      rs += __shfl_xor(rs, 2, 32);
      rs += __shfl_xor(rs, 4, 32);
      rs += __shfl_xor(rs, 8, 32);
      lrow[e] = lrow[e] * sc + rs;
      O0[e] *= sc; O1[e] *= sc;
      x0[e] = p0; x1[e] = p1;
    }
#pragma unroll
    for (int e = 0; e < 8; ++e) {  // C-layout -> A-layout via per-wave LDS tile
      Pl[wv][hs8 + e][cl]      = (_Float16)x0[e];
      Pl[wv][hs8 + e][cl + 16] = (_Float16)x1[e];
    }
    v16h pf  = frag_a(&Pl[wv][0][0], 32, lane);
    v16h vb0 = frag_b(&Vl[0][col0],  352, lane);
    v16h vb1 = frag_b(&Vl[16][col0], 352, lane);
    O0 = __builtin_amdgcn_wmma_f32_16x16x32_f16(false, pf, false, vb0, (short)0, O0, false, false);
    O1 = __builtin_amdgcn_wmma_f32_16x16x32_f16(false, pf, false, vb1, (short)0, O1, false, false);
  }
#pragma unroll
  for (int e = 0; e < 8; ++e) {
    int r = qrow0 + hs8 + e;
    if (r < NTOK) {
      float inv = 1.0f / lrow[e];
      size_t row = (size_t)win * NTOK + r;
      Ob[row * DIM + head * 32 + cl]      = (_Float16)(O0[e] * inv);
      Ob[row * DIM + head * 32 + cl + 16] = (_Float16)(O1[e] * inv);
    }
  }
}

// ---------- fused elementwise kernels ----------
__global__ void __launch_bounds__(192)
fuse1_kernel(const float* __restrict__ proj, const float* __restrict__ x,
             const float* __restrict__ g, const float* __restrict__ b,
             float* __restrict__ x2, _Float16* __restrict__ x2h) {
  __shared__ float red[8];
  __shared__ float red2[8];
  int s = blockIdx.x, ch = threadIdx.x;
  int d = s / 3136, rem = s - d * 3136, hh = rem / 56, ww = rem - hh * 56;
  int dp = (d + 11) % 14, hp = (hh + 53) % 56, wp = (ww + 53) % 56; // inverse roll(+3)
  int win = ((dp / 7) * 8 + hp / 7) * 8 + wp / 7;
  int token = ((dp % 7) * 7 + hp % 7) * 7 + (wp % 7);
  float v = proj[((size_t)win * NTOK + token) * DIM + ch];
  int lane = ch & 31, wv = ch >> 5;
  float sum = v;
  sum += __shfl_xor(sum, 1, 32); sum += __shfl_xor(sum, 2, 32);
  sum += __shfl_xor(sum, 4, 32); sum += __shfl_xor(sum, 8, 32);
  sum += __shfl_xor(sum, 16, 32);
  if (lane == 0) red[wv] = sum;
  __syncthreads();
  float mean = (red[0] + red[1] + red[2] + red[3] + red[4] + red[5]) * (1.0f / DIM);
  float dv = v - mean, sq = dv * dv;
  sq += __shfl_xor(sq, 1, 32); sq += __shfl_xor(sq, 2, 32);
  sq += __shfl_xor(sq, 4, 32); sq += __shfl_xor(sq, 8, 32);
  sq += __shfl_xor(sq, 16, 32);
  if (lane == 0) red2[wv] = sq;
  __syncthreads();
  float var = (red2[0] + red2[1] + red2[2] + red2[3] + red2[4] + red2[5]) * (1.0f / DIM);
  float y = dv * rsqrtf(var + 1e-5f) * g[ch] + b[ch];
  float outv = x[(size_t)s * DIM + ch] + y;
  x2[(size_t)s * DIM + ch] = outv;
  x2h[(size_t)s * DIM + ch] = (_Float16)outv;
}

__global__ void __launch_bounds__(192)
fuse2_kernel(const float* __restrict__ mlp, const float* __restrict__ x2,
             const float* __restrict__ g, const float* __restrict__ b,
             const float* __restrict__ clin, const float* __restrict__ cw,
             const float* __restrict__ cb, float* __restrict__ out) {
  __shared__ float red[8];
  __shared__ float red2[8];
  int s = blockIdx.x, ch = threadIdx.x;
  float v = mlp[(size_t)s * DIM + ch];
  int lane = ch & 31, wv = ch >> 5;
  float sum = v;
  sum += __shfl_xor(sum, 1, 32); sum += __shfl_xor(sum, 2, 32);
  sum += __shfl_xor(sum, 4, 32); sum += __shfl_xor(sum, 8, 32);
  sum += __shfl_xor(sum, 16, 32);
  if (lane == 0) red[wv] = sum;
  __syncthreads();
  float mean = (red[0] + red[1] + red[2] + red[3] + red[4] + red[5]) * (1.0f / DIM);
  float dv = v - mean, sq = dv * dv;
  sq += __shfl_xor(sq, 1, 32); sq += __shfl_xor(sq, 2, 32);
  sq += __shfl_xor(sq, 4, 32); sq += __shfl_xor(sq, 8, 32);
  sq += __shfl_xor(sq, 16, 32);
  if (lane == 0) red2[wv] = sq;
  __syncthreads();
  float var = (red2[0] + red2[1] + red2[2] + red2[3] + red2[4] + red2[5]) * (1.0f / DIM);
  float y = dv * rsqrtf(var + 1e-5f) * g[ch] + b[ch];
  float clv = clin[0] * cw[ch * 3] + clin[1] * cw[ch * 3 + 1] +
              clin[2] * cw[ch * 3 + 2] + cb[ch];
  out[(size_t)s * DIM + ch] = x2[(size_t)s * DIM + ch] + y + clv;
}

// ---------- small prep kernels ----------
__global__ void cvt_f16_kernel(const float* __restrict__ src, _Float16* __restrict__ dst, int n) {
  int i = blockIdx.x * blockDim.x + threadIdx.x;
  if (i < n) dst[i] = (_Float16)src[i];
}

__global__ void gather_win_kernel(const float* __restrict__ x, _Float16* __restrict__ winA) {
  int i = blockIdx.x * blockDim.x + threadIdx.x;
  if (i >= SITES * DIM) return;
  int row = i / DIM, ch = i - row * DIM;
  int win = row / NTOK, token = row - win * NTOK;
  int wd = win / 64, r1 = win - wd * 64, wh_ = r1 >> 3, ww_ = r1 & 7;
  int td = token / 49, r2 = token - td * 49, th = r2 / 7, tw = r2 - th * 7;
  int d = (wd * 7 + td + 3) % 14;   // forward roll(-3)
  int h = (wh_ * 7 + th + 3) % 56;
  int w = (ww_ * 7 + tw + 3) % 56;
  winA[(size_t)row * DIM + ch] = (_Float16)x[((size_t)(d * 56 + h) * 56 + w) * DIM + ch];
}

__global__ void bias_pre_kernel(const float* __restrict__ tbl, const int* __restrict__ ridx,
                                float* __restrict__ biasH) {
  int i = blockIdx.x * blockDim.x + threadIdx.x;
  if (i >= NTOK * NTOK) return;
  int id = ridx[i];
#pragma unroll
  for (int h = 0; h < HEADS; ++h)
    biasH[(size_t)h * NTOK * NTOK + i] = tbl[id * HEADS + h];
}

// ---------- host launcher ----------
extern "C" void kernel_launch(void* const* d_in, const int* in_sizes, int n_in,
                              void* d_out, int out_size, void* d_ws, size_t ws_size,
                              hipStream_t stream) {
  (void)in_sizes; (void)n_in; (void)out_size; (void)ws_size;
  const float* x      = (const float*)d_in[0];
  const float* mask   = (const float*)d_in[1];
  const float* clin   = (const float*)d_in[2];
  const float* qkv_w  = (const float*)d_in[3];
  const float* qkv_b  = (const float*)d_in[4];
  const float* proj_w = (const float*)d_in[5];
  const float* proj_b = (const float*)d_in[6];
  const float* tbl    = (const float*)d_in[7];
  const float* n1g    = (const float*)d_in[8];
  const float* n1b    = (const float*)d_in[9];
  const float* n2g    = (const float*)d_in[10];
  const float* n2b    = (const float*)d_in[11];
  const float* f1w    = (const float*)d_in[12];
  const float* f1b    = (const float*)d_in[13];
  const float* f2w    = (const float*)d_in[14];
  const float* f2b    = (const float*)d_in[15];
  const float* cw     = (const float*)d_in[16];
  const float* cb     = (const float*)d_in[17];
  const int*   ridx   = (const int*)d_in[18];
  float* out = (float*)d_out;

  char* ws = (char*)d_ws;
  size_t off = 0;
  auto alloc = [&](size_t bytes) {
    char* p = ws + off;
    off = (off + bytes + 255) & ~(size_t)255;
    return p;
  };
  _Float16* wq16 = (_Float16*)alloc((size_t)576 * 192 * 2);
  _Float16* wp16 = (_Float16*)alloc((size_t)192 * 192 * 2);
  _Float16* w116 = (_Float16*)alloc((size_t)768 * 192 * 2);
  _Float16* w216 = (_Float16*)alloc((size_t)192 * 768 * 2);
  float*    biasH = (float*)alloc((size_t)HEADS * NTOK * NTOK * 4);
  _Float16* winA = (_Float16*)alloc((size_t)SITES * DIM * 2);   // reused as attn_out
  _Float16* qb = (_Float16*)alloc((size_t)NWIN * HEADS * NPAD * 32 * 2);
  _Float16* kb = (_Float16*)alloc((size_t)NWIN * HEADS * NPAD * 32 * 2);
  _Float16* vb = (_Float16*)alloc((size_t)NWIN * HEADS * NPAD * 32 * 2);
  float*    projf = (float*)alloc((size_t)SITES * DIM * 4);     // reused as mlp out
  float*    x2   = (float*)alloc((size_t)SITES * DIM * 4);
  _Float16* x2h  = (_Float16*)alloc((size_t)SITES * DIM * 2);
  _Float16* h1   = (_Float16*)alloc((size_t)SITES * MLPH * 2);

  cvt_f16_kernel<<<(576 * 192 + 255) / 256, 256, 0, stream>>>(qkv_w, wq16, 576 * 192);
  cvt_f16_kernel<<<(192 * 192 + 255) / 256, 256, 0, stream>>>(proj_w, wp16, 192 * 192);
  cvt_f16_kernel<<<(768 * 192 + 255) / 256, 256, 0, stream>>>(f1w, w116, 768 * 192);
  cvt_f16_kernel<<<(192 * 768 + 255) / 256, 256, 0, stream>>>(f2w, w216, 192 * 768);
  bias_pre_kernel<<<(NTOK * NTOK + 255) / 256, 256, 0, stream>>>(tbl, ridx, biasH);
  gather_win_kernel<<<(SITES * DIM + 255) / 256, 256, 0, stream>>>(x, winA);

  gemm_kernel<0><<<dim3(576 / 64, SITES / 128), 256, 0, stream>>>(
      winA, wq16, qkv_b, nullptr, nullptr, qb, kb, vb, SITES, 576, 192);
  attn_kernel<<<dim3(NWIN * HEADS, 3), 256, 0, stream>>>(qb, kb, vb, biasH, mask, winA);
  gemm_kernel<1><<<dim3(192 / 64, SITES / 128), 256, 0, stream>>>(
      winA, wp16, proj_b, projf, nullptr, nullptr, nullptr, nullptr, SITES, 192, 192);
  fuse1_kernel<<<SITES, 192, 0, stream>>>(projf, x, n1g, n1b, x2, x2h);
  gemm_kernel<2><<<dim3(768 / 64, SITES / 128), 256, 0, stream>>>(
      x2h, w116, f1b, nullptr, h1, nullptr, nullptr, nullptr, SITES, 768, 192);
  gemm_kernel<3><<<dim3(192 / 64, SITES / 128), 256, 0, stream>>>(
      h1, w216, f2b, projf, nullptr, nullptr, nullptr, nullptr, SITES, 192, 768);
  fuse2_kernel<<<SITES, 192, 0, stream>>>(projf, x2, n2g, n2b, clin, cw, cb, out);
}
